// MambaBlock_515396075671
// MI455X (gfx1250) — matrix operand order repaired and checked by reading, not compiled, not run on hardware
//
#include <hip/hip_runtime.h>
#include <hip/hip_bf16.h>
#include <math.h>

// ---------------------------------------------------------------------------
// Mamba block forward for MI455X (gfx1250), wave32 + WMMA bf16.
// B=2, S=2048, DIM=768, INNER=1536, STATE=16, K=4.
// ---------------------------------------------------------------------------

#define BATCH 2
#define SEQ   2048
#define DIM_  768
#define INNER 1536
#define STATE 16
#define KCONV 4
#define MROWS (BATCH * SEQ)     // 4096
#define N_XZ  (2 * INNER)       // 3072

typedef __attribute__((ext_vector_type(16))) __bf16 v16bf;
typedef __attribute__((ext_vector_type(8)))  float  v8f;

union Frag16 { v16bf v; unsigned u[8]; };
union Frag32 { v8f   v; float    f[8]; };

// f32 pair -> packed bf16 dword (backend picks native cvt, RNE)
__device__ __forceinline__ unsigned pk_bf16(float a, float b) {
    union { __bf16 h[2]; unsigned u; } p;
    p.h[0] = (__bf16)a;
    p.h[1] = (__bf16)b;
    return p.u;
}

__device__ __forceinline__ float silu_f(float v) {
    return v / (1.0f + __expf(-v));
}

// ---------------------------------------------------------------------------
// 1. RMSNorm: one block per (b,s) row of 768
// ---------------------------------------------------------------------------
__global__ __launch_bounds__(256) void rmsnorm_kernel(const float* __restrict__ x,
                                                      const float* __restrict__ w,
                                                      float* __restrict__ xn) {
    __shared__ float red[256];
    const int row = blockIdx.x;
    const int tid = threadIdx.x;
    const float* xr = x + (size_t)row * DIM_;
    float ss = 0.0f;
    for (int j = tid; j < DIM_; j += 256) { float v = xr[j]; ss += v * v; }
    red[tid] = ss;
    __syncthreads();
    for (int off = 128; off > 0; off >>= 1) {
        if (tid < off) red[tid] += red[tid + off];
        __syncthreads();
    }
    const float scale = rsqrtf(red[0] * (1.0f / DIM_) + 1e-6f);
    for (int j = tid; j < DIM_; j += 256)
        xn[(size_t)row * DIM_ + j] = xr[j] * scale * w[j];
}

// ---------------------------------------------------------------------------
// 2. WMMA GEMM: C[M x N] = A[M x Kd] * W[N x Kd]^T  (+bias/epilogue by MODE)
//    f32 in global, bf16 staged in LDS, fp32 accumulate via
//    v_wmma_f32_16x16x32_bf16. Block tile 64(M) x 128(N), K-chunk 32,
//    register double-buffer so global loads of chunk k+1 overlap the WMMAs
//    of chunk k. 8 waves: wave w -> M-tile (w&3), N-tiles (w>>2)*4 .. +3.
//    MODE 0: plain store; MODE 1: softplus(acc + bias)
// ---------------------------------------------------------------------------
#define LDT 40   // LDS tile stride in halves (20-bank lane stride, conflict-free)

template <int MODE>
__global__ __launch_bounds__(256) void gemm_bf16_wmma_kernel(
        const float* __restrict__ A, const float* __restrict__ W,
        const float* __restrict__ bias, float* __restrict__ Cout,
        int Kd, int N) {
    __shared__ unsigned short lAs[64  * LDT];
    __shared__ unsigned short lWs[128 * LDT];

    const int tid    = threadIdx.x;
    const int lane   = tid & 31;
    const int wave   = tid >> 5;
    const int mBlock = blockIdx.x * 64;
    const int nBlock = blockIdx.y * 128;
    const int mLoc   = (wave & 3) * 16;        // wave's M-tile inside block
    const int nLoc0  = (wave >> 2) * 64;       // wave's first N-tile inside block

    // per-thread staging coordinates (8 float4 per 32-wide K chunk per row)
    const int arow0 = tid >> 3;                // A rows for it=0/1: arow0, arow0+32
    const int acv   = tid & 7;
    const size_t aOff = (size_t)(mBlock + arow0) * Kd + acv * 4;
    const size_t wOff = (size_t)(nBlock + arow0) * Kd + acv * 4;

    Frag32 c[4];
#pragma unroll
    for (int q = 0; q < 4; ++q)
#pragma unroll
        for (int r = 0; r < 8; ++r) c[q].f[r] = 0.0f;

    // ---- prefetch chunk 0 into registers
    float4 pa[2], pw[4];
#pragma unroll
    for (int it = 0; it < 2; ++it)
        pa[it] = *(const float4*)(A + aOff + (size_t)it * 32 * Kd);
#pragma unroll
    for (int it = 0; it < 4; ++it)
        pw[it] = *(const float4*)(W + wOff + (size_t)it * 32 * Kd);

    for (int kc = 0; kc < Kd; kc += 32) {
        __syncthreads();   // previous iteration's LDS reads complete
        // ---- convert + store staged registers into LDS (bf16)
#pragma unroll
        for (int it = 0; it < 2; ++it) {
            unsigned* dst = (unsigned*)&lAs[(arow0 + it * 32) * LDT + acv * 4];
            dst[0] = pk_bf16(pa[it].x, pa[it].y);
            dst[1] = pk_bf16(pa[it].z, pa[it].w);
        }
#pragma unroll
        for (int it = 0; it < 4; ++it) {
            unsigned* dst = (unsigned*)&lWs[(arow0 + it * 32) * LDT + acv * 4];
            dst[0] = pk_bf16(pw[it].x, pw[it].y);
            dst[1] = pk_bf16(pw[it].z, pw[it].w);
        }
        __syncthreads();

        // ---- issue global loads for chunk kc+32 (latency hidden by WMMAs)
        if (kc + 32 < Kd) {
            const size_t koff = (size_t)(kc + 32);
#pragma unroll
            for (int it = 0; it < 2; ++it)
                pa[it] = *(const float4*)(A + aOff + (size_t)it * 32 * Kd + koff);
#pragma unroll
            for (int it = 0; it < 4; ++it)
                pw[it] = *(const float4*)(W + wOff + (size_t)it * 32 * Kd + koff);
        }

        // ---- A fragment (16-bit A 16x32 layout):
        // lane L: m = L%16; uint v holds K pair at K0 = (v/4)*16 + (L/16)*8 + (v%4)*2
        Frag16 fa;
        {
            const int m  = mLoc + (lane & 15);
            const int hi = lane >> 4;
            const unsigned* src = (const unsigned*)&lAs[m * LDT];
#pragma unroll
            for (int v = 0; v < 8; ++v) {
                const int K0 = ((v >> 2) << 4) + (hi << 3) + ((v & 3) << 1);
                fa.u[v] = src[K0 >> 1];
            }
        }
        // ---- B fragments (32x16 layout): lane L: n = L%16; uint v holds K pair at K0 = (v/4)*16 + (L/16)*8 + (v%4)*2
        // B[k][n] = W[n][k] -> read row n of the W tile
#pragma unroll
        for (int q = 0; q < 4; ++q) {
            Frag16 fb;
            const int n  = nLoc0 + q * 16 + (lane & 15);
            const int hi = lane >> 4;
            const unsigned* src = (const unsigned*)&lWs[n * LDT];
#pragma unroll
            for (int v = 0; v < 8; ++v) {
                const int K0 = ((v >> 2) << 4) + (hi << 3) + ((v & 3) << 1);
                fb.u[v] = src[K0 >> 1];
            }
            c[q].v = __builtin_amdgcn_wmma_f32_16x16x32_bf16(
                false, fa.v, false, fb.v, (short)0, c[q].v, false, false);
        }
    }

    // ---- epilogue: C/D layout — VGPR r: lanes 0-15 -> M=r, lanes 16-31 -> M=8+r
    const int rowBase = mBlock + mLoc + (lane >> 4) * 8;
    const int colBase = nBlock + nLoc0 + (lane & 15);
#pragma unroll
    for (int q = 0; q < 4; ++q) {
        const int n = colBase + q * 16;
        float bv = 0.0f;
        if (MODE == 1) bv = bias[n];
#pragma unroll
        for (int r = 0; r < 8; ++r) {
            float v = c[q].f[r] + bv;
            if (MODE == 1) v = (v > 20.0f) ? v : log1pf(__expf(v));   // softplus
            Cout[(size_t)(rowBase + r) * N + n] = v;
        }
    }
}

// ---------------------------------------------------------------------------
// 3. Depthwise causal conv (K=4) + bias + SiLU. x_inner lives inside xz.
// ---------------------------------------------------------------------------
__global__ __launch_bounds__(256) void conv_silu_kernel(const float* __restrict__ xz,
                                                        const float* __restrict__ cw,
                                                        const float* __restrict__ cb,
                                                        float* __restrict__ xc) {
    const int idx = blockIdx.x * 256 + threadIdx.x;
    if (idx >= MROWS * INNER) return;
    const int i  = idx % INNER;
    const int rs = idx / INNER;          // b*SEQ + s
    const int s  = rs % SEQ;
    float acc = cb[i];
#pragma unroll
    for (int t = 0; t < KCONV; ++t) {
        const int ss = s - (KCONV - 1) + t;
        if (ss >= 0)
            acc += xz[(size_t)(rs - (KCONV - 1) + t) * N_XZ + i] * cw[i * KCONV + t];
    }
    xc[idx] = silu_f(acc);
}

// ---------------------------------------------------------------------------
// 4. B/C projections (N=16 each): one wave per (b,s) row; lanes 0-15 -> Bm,
//    lanes 16-31 -> Cm. xc reads are wave-uniform; weights stay cache-resident.
// ---------------------------------------------------------------------------
__global__ __launch_bounds__(256) void bc_proj_kernel(const float* __restrict__ xc,
                                                      const float* __restrict__ Bw,
                                                      const float* __restrict__ Cw,
                                                      float* __restrict__ Bm,
                                                      float* __restrict__ Cm) {
    const int wave = threadIdx.x >> 5;
    const int lane = threadIdx.x & 31;
    const int row  = blockIdx.x * 8 + wave;
    const int n    = lane & 15;
    const float* Wp = (lane < 16) ? Bw : Cw;
    const float* xr = xc + (size_t)row * INNER;
    const float* wr = Wp + (size_t)n * INNER;
    float acc = 0.0f;
    for (int k = 0; k < INNER; k += 4) {
        const float4 xv = *(const float4*)(xr + k);
        const float4 wv = *(const float4*)(wr + k);
        acc += xv.x * wv.x + xv.y * wv.y + xv.z * wv.z + xv.w * wv.w;
    }
    float* dst = (lane < 16) ? Bm : Cm;
    dst[(size_t)row * STATE + n] = acc;
}

// ---------------------------------------------------------------------------
// 5. Selective scan: one thread per (b, channel); h[16] + A-row in registers.
//    B/C loads are uniform across each block (one batch per block range).
// ---------------------------------------------------------------------------
__global__ __launch_bounds__(256) void scan_kernel(const float* __restrict__ dt,
                                                   const float* __restrict__ xc,
                                                   const float* __restrict__ Bm,
                                                   const float* __restrict__ Cm,
                                                   const float* __restrict__ A_log,
                                                   const float* __restrict__ D,
                                                   float* __restrict__ y) {
    const int g = blockIdx.x * 256 + threadIdx.x;   // 0 .. B*INNER-1
    const int b = g / INNER;
    const int i = g % INNER;
    float a[STATE], h[STATE];
#pragma unroll
    for (int n = 0; n < STATE; ++n) {
        a[n] = -__expf(A_log[(size_t)i * STATE + n]);
        h[n] = 0.0f;
    }
    const float Dv = D[i];
    const size_t base = (size_t)b * SEQ;
    for (int s = 0; s < SEQ; ++s) {
        const size_t r   = base + s;
        const float dtv  = dt[r * INNER + i];
        const float xcv  = xc[r * INNER + i];
        const float dtx  = dtv * xcv;
        float acc = Dv * xcv;
        const float* Bp = Bm + r * STATE;
        const float* Cp = Cm + r * STATE;
#pragma unroll
        for (int n = 0; n < STATE; ++n) {
            const float dA = __expf(a[n] * dtv);
            h[n] = fmaf(dA, h[n], dtx * Bp[n]);
            acc  = fmaf(h[n], Cp[n], acc);
        }
        y[r * INNER + i] = acc;
    }
}

// ---------------------------------------------------------------------------
// 6. Gate: y *= silu(res)   (res lives in the second half of xz rows)
// ---------------------------------------------------------------------------
__global__ __launch_bounds__(256) void gate_kernel(const float* __restrict__ xz,
                                                   float* __restrict__ y) {
    const int idx = blockIdx.x * 256 + threadIdx.x;
    if (idx >= MROWS * INNER) return;
    const int i  = idx % INNER;
    const int rs = idx / INNER;
    const float res = xz[(size_t)rs * N_XZ + INNER + i];
    y[idx] = y[idx] * silu_f(res);
}

// ---------------------------------------------------------------------------
// launch
// ---------------------------------------------------------------------------
extern "C" void kernel_launch(void* const* d_in, const int* in_sizes, int n_in,
                              void* d_out, int out_size, void* d_ws, size_t ws_size,
                              hipStream_t stream) {
    (void)in_sizes; (void)n_in; (void)out_size; (void)ws_size;
    const float* x     = (const float*)d_in[0];
    const float* normw = (const float*)d_in[1];
    const float* inpW  = (const float*)d_in[2];
    const float* convW = (const float*)d_in[3];
    const float* convB = (const float*)d_in[4];
    const float* dtW   = (const float*)d_in[5];
    const float* dtB   = (const float*)d_in[6];
    const float* Bw    = (const float*)d_in[7];
    const float* Cw    = (const float*)d_in[8];
    const float* Alog  = (const float*)d_in[9];
    const float* Dp    = (const float*)d_in[10];
    const float* outW  = (const float*)d_in[11];
    float* out = (float*)d_out;

    float* ws = (float*)d_ws;
    float* xn  = ws;                                  // 4096*768
    float* xz  = xn  + (size_t)MROWS * DIM_;          // 4096*3072
    float* xc  = xz  + (size_t)MROWS * N_XZ;          // 4096*1536
    float* dtb = xc  + (size_t)MROWS * INNER;         // 4096*1536
    float* Bm  = dtb + (size_t)MROWS * INNER;         // 4096*16
    float* Cm  = Bm  + (size_t)MROWS * STATE;         // 4096*16
    float* y   = Cm  + (size_t)MROWS * STATE;         // 4096*1536

    const int elemBlocks = (MROWS * INNER) / 256;     // 24576

    // 1. RMSNorm
    rmsnorm_kernel<<<MROWS, 256, 0, stream>>>(x, normw, xn);
    // 2. in_proj GEMM: 4096x768 * (3072x768)^T -> xz
    gemm_bf16_wmma_kernel<0><<<dim3(MROWS / 64, N_XZ / 128), 256, 0, stream>>>(
        xn, inpW, nullptr, xz, DIM_, N_XZ);
    // 3. depthwise conv + SiLU -> xc
    conv_silu_kernel<<<elemBlocks, 256, 0, stream>>>(xz, convW, convB, xc);
    // 4. dt GEMM + softplus(.+dt_b): 4096x1536 * (1536x1536)^T -> dtb
    gemm_bf16_wmma_kernel<1><<<dim3(MROWS / 64, INNER / 128), 256, 0, stream>>>(
        xc, dtW, dtB, dtb, INNER, INNER);
    // 5. B/C projections
    bc_proj_kernel<<<MROWS / 8, 256, 0, stream>>>(xc, Bw, Cw, Bm, Cm);
    // 6. selective scan
    scan_kernel<<<(BATCH * INNER) / 256, 256, 0, stream>>>(dtb, xc, Bm, Cm, Alog, Dp, y);
    // 7. gate: y *= silu(res)
    gate_kernel<<<elemBlocks, 256, 0, stream>>>(xz, y);
    // 8. out GEMM: 4096x1536 * (768x1536)^T -> out
    gemm_bf16_wmma_kernel<0><<<dim3(MROWS / 64, DIM_ / 128), 256, 0, stream>>>(
        y, outW, nullptr, out, INNER, DIM_);
}
